// FourierCrossAttentionW_45423574122660
// MI455X (gfx1250) — matrix operand order repaired and checked by reading, not compiled, not run on hardware
//
#include <hip/hip_runtime.h>
#include <hip/hip_bf16.h>
#include <math.h>

typedef __attribute__((ext_vector_type(16))) __bf16 v16bf;
typedef __attribute__((ext_vector_type(8)))  float  v8f;
typedef __attribute__((ext_vector_type(2)))  float  v2f;

#define BATCH 16
#define HEADS 8
#define SEQ   2048
#define EDIM  64
#define MODES2 128               // 64 real cols | 64 imag cols, packed
#define BH    (BATCH*HEADS)      // 128
#define FT_STRIDE (EDIM*MODES2)  // 8192 floats per (b,h)

static __device__ inline v8f wmma_bf16(v16bf a, v16bf b, v8f c) {
  return __builtin_amdgcn_wmma_f32_16x16x32_bf16(false, a, false, b, (short)0, c,
                                                 false, false);
}
static __device__ inline v8f wmma_f32(v2f a, v2f b, v8f c) {
  return __builtin_amdgcn_wmma_f32_16x16x4_f32(false, a, false, b, (short)0, c,
                                               false, false);
}
static __device__ inline __bf16 f2bf(float f) { return (__bf16)f; }

// A-matrix (bf16, 16x32) element k-index for vector element i, given lane half.
// Elements 0..7 -> k = 8*half + 0..7 ; elements 8..15 -> k = 16 + 8*half + 0..7
// (two contiguous 16B runs -> ds_load_b128 pairs).
static __device__ inline int a16x32_k(int i, int half) {
  int j = i >> 1, lo = i & 1;
  return (j < 4) ? (2 * j + 8 * half + lo) : (16 + 2 * (j - 4) + 8 * half + lo);
}

// ---------------------------------------------------------------------------
// Twiddles, stored K-contiguous for the consuming GEMMs:
//   WfwdT[m'][t] (fwd DFT B-matrix, K = t contiguous)
//   WinvT[t][m'] (inv DFT B-matrix, K = m' contiguous), scale 1/(N*64*64) and
//   irfft coefficients (c_0=1, c_m=2) folded in. Angle via (m*t) mod 2048.
// ---------------------------------------------------------------------------
__global__ __launch_bounds__(256) void twiddle_init(__bf16* __restrict__ WfwdT,
                                                    __bf16* __restrict__ WinvT) {
  int idx = blockIdx.x * blockDim.x + threadIdx.x;   // 0 .. 262143
  if (idx >= SEQ * MODES2) return;
  int t  = idx >> 7;
  int mp = idx & 127;
  int m  = mp & 63;
  int phase = (m * t) & (SEQ - 1);
  float theta = 6.283185307179586f * (float)phase * (1.0f / (float)SEQ);
  float s, c;
  __sincosf(theta, &s, &c);
  WfwdT[(size_t)mp * SEQ + t] = f2bf((mp < 64) ? c : -s);
  float cm  = (m == 0) ? 1.0f : 2.0f;
  float inv = cm * (1.0f / ((float)SEQ * 4096.0f));
  WinvT[idx] = f2bf((mp < 64) ? (c * inv) : (-s * inv));   // idx == t*128+mp
}

// ---------------------------------------------------------------------------
// Forward truncated DFT. Rows = flattened (e*8+h) so A chunks are contiguous
// segments of q/k (coalesced float4 loads). grid.x = 2*16*8 = 256.
// Each WG: C[64 rows x 128 m'], K = 2048 in 64 chunks of 32, double-buffered.
// ---------------------------------------------------------------------------
__global__ __launch_bounds__(256) void dft_fwd(const float* __restrict__ q,
                                               const float* __restrict__ k,
                                               const __bf16* __restrict__ WfwdT,
                                               float* __restrict__ ft) {
  __shared__ __align__(16) __bf16 As[2][64 * 40];    // [row][t] stride 40
  __shared__ __align__(16) __bf16 Bs[2][128 * 40];   // [m'][t]  stride 40

  int wg    = blockIdx.x;        // 0..255
  int which = wg >> 7;           // 0=q, 1=k
  int rem   = wg & 127;
  int b = rem >> 3, rb = rem & 7;                 // rb: 64-row block of 512
  const float* src = (which ? k : q) + (size_t)b * SEQ * 512 + rb * 64;
  float* dstbase = ft + (size_t)(which * BH + b * HEADS) * FT_STRIDE;

  int tid  = threadIdx.x;
  int lane = tid & 31, wave = tid >> 5;
  int mrow = lane & 15, half = lane >> 4;
  int rt = wave & 3;             // 4 row tiles, 1 per wave
  int c0 = (wave >> 2) * 4;      // 8 col tiles, 4 per wave

  // Loader mappings: A = 512 float4 (2/thread), B = 512 uint4 (2/thread)
  int tA[2], rqA[2], rowB[2], cqB[2];
#pragma unroll
  for (int v = 0; v < 2; ++v) {
    int lin = v * 256 + tid;
    tA[v]   = lin >> 4;          // 16 float4 per t-row of 64 floats
    rqA[v]  = lin & 15;
    rowB[v] = lin >> 2;          // 4 uint4 per m'-row of 32 bf16
    cqB[v]  = lin & 3;
  }

  v8f acc[4];
#pragma unroll
  for (int cj = 0; cj < 4; ++cj)
#pragma unroll
    for (int r = 0; r < 8; ++r) acc[cj][r] = 0.0f;

  float4 fa[2];
  uint4  fb[2];
  // Prologue: fetch + stage chunk 0 into buffer 0.
#pragma unroll
  for (int v = 0; v < 2; ++v) {
    fa[v] = *(const float4*)(src + (size_t)tA[v] * 512 + rqA[v] * 4);
    fb[v] = *(const uint4*)(WfwdT + (size_t)rowB[v] * SEQ + cqB[v] * 8);
  }
#pragma unroll
  for (int v = 0; v < 2; ++v) {
    As[0][(rqA[v] * 4 + 0) * 40 + tA[v]] = f2bf(fa[v].x);
    As[0][(rqA[v] * 4 + 1) * 40 + tA[v]] = f2bf(fa[v].y);
    As[0][(rqA[v] * 4 + 2) * 40 + tA[v]] = f2bf(fa[v].z);
    As[0][(rqA[v] * 4 + 3) * 40 + tA[v]] = f2bf(fa[v].w);
    *(uint4*)(&Bs[0][rowB[v] * 40 + cqB[v] * 8]) = fb[v];
  }
  __syncthreads();

#pragma unroll 1
  for (int kt = 0; kt < 64; ++kt) {
    int cur = kt & 1, nxt = cur ^ 1;
    bool more = (kt + 1) < 64;
    // Issue next chunk's global loads before compute (latency overlap).
    if (more) {
      int t0 = (kt + 1) * 32;
#pragma unroll
      for (int v = 0; v < 2; ++v) {
        fa[v] = *(const float4*)(src + (size_t)(t0 + tA[v]) * 512 + rqA[v] * 4);
        fb[v] = *(const uint4*)(WfwdT + (size_t)rowB[v] * SEQ + t0 + cqB[v] * 8);
      }
    }
    // Compute on current buffer.
    v16bf a;
#pragma unroll
    for (int i = 0; i < 16; ++i)
      a[i] = As[cur][(rt * 16 + mrow) * 40 + a16x32_k(i, half)];
#pragma unroll
    for (int cj = 0; cj < 4; ++cj) {
      v16bf bfrag;
#pragma unroll
      for (int i = 0; i < 16; ++i)
        bfrag[i] = Bs[cur][((c0 + cj) * 16 + mrow) * 40 + a16x32_k(i, half)];
      acc[cj] = wmma_bf16(a, bfrag, acc[cj]);
    }
    // Stage next chunk into the other buffer.
    if (more) {
#pragma unroll
      for (int v = 0; v < 2; ++v) {
        As[nxt][(rqA[v] * 4 + 0) * 40 + tA[v]] = f2bf(fa[v].x);
        As[nxt][(rqA[v] * 4 + 1) * 40 + tA[v]] = f2bf(fa[v].y);
        As[nxt][(rqA[v] * 4 + 2) * 40 + tA[v]] = f2bf(fa[v].z);
        As[nxt][(rqA[v] * 4 + 3) * 40 + tA[v]] = f2bf(fa[v].w);
        *(uint4*)(&Bs[nxt][rowB[v] * 40 + cqB[v] * 8]) = fb[v];
      }
    }
    __syncthreads();
  }

  // Epilogue: row R = rb*64 + rt*16 + r + 8*half -> e = R>>3, h = r.
  int e = rb * 8 + rt * 2 + half;
#pragma unroll
  for (int cj = 0; cj < 4; ++cj) {
    int n = (c0 + cj) * 16 + mrow;
#pragma unroll
    for (int r = 0; r < 8; ++r)
      dstbase[(size_t)r * FT_STRIDE + e * MODES2 + n] = acc[cj][r];
  }
}

// ---------------------------------------------------------------------------
// Mode-space complex GEMM #1 + tanh:  T[x,y] = tanh( sum_e qft[e,x]*kft[e,y] )
// f32 WMMA 16x16x4; -Aim applied in VALU (F32 WMMA has no A/B NEG per ISA).
// grid.x = 2*BH, one 16x16 complex tile per wave.
// ---------------------------------------------------------------------------
__global__ __launch_bounds__(256) void qk_tanh(const float* __restrict__ qft,
                                               const float* __restrict__ kft,
                                               float* __restrict__ T) {
  int bh  = blockIdx.x >> 1;
  int sub = blockIdx.x & 1;
  const float* Q = qft + (size_t)bh * FT_STRIDE;
  const float* K = kft + (size_t)bh * FT_STRIDE;
  float* Tre = T + (size_t)bh * 8192;
  float* Tim = Tre + 4096;

  int tid = threadIdx.x, lane = tid & 31, wave = tid >> 5;
  int mrow = lane & 15, half = lane >> 4;

  int tile = wave + 8 * sub;      // 16 tiles (4x4), 1 per wave
  int xi = tile >> 2, yi = tile & 3;
  v8f accR, accI;
#pragma unroll
  for (int r = 0; r < 8; ++r) { accR[r] = 0.0f; accI[r] = 0.0f; }

  for (int e0 = 0; e0 < 64; e0 += 4) {
    v2f aR, aI, aIn, bR, bI;
#pragma unroll
    for (int j = 0; j < 2; ++j) {
      int ke = e0 + j + 2 * half;
      aR[j] = Q[ke * MODES2 + (xi * 16 + mrow)];
      aI[j] = Q[ke * MODES2 + 64 + (xi * 16 + mrow)];
      bR[j] = K[ke * MODES2 + (yi * 16 + mrow)];
      bI[j] = K[ke * MODES2 + 64 + (yi * 16 + mrow)];
      aIn[j] = -aI[j];
    }
    accR = wmma_f32(aR,  bR, accR);
    accR = wmma_f32(aIn, bI, accR);
    accI = wmma_f32(aR,  bI, accI);
    accI = wmma_f32(aI,  bR, accI);
  }
#pragma unroll
  for (int r = 0; r < 8; ++r) {
    int x = xi * 16 + r + 8 * half;
    int y = yi * 16 + mrow;
    Tre[x * 64 + y] = tanhf(accR[r]);
    Tim[x * 64 + y] = tanhf(accI[r]);
  }
}

// ---------------------------------------------------------------------------
// Mode-space complex GEMM #2: X2[e,x] = sum_y T[x,y] * kft[e,y]
// grid.x = 2*BH, one 16x16 complex tile per wave.
// ---------------------------------------------------------------------------
__global__ __launch_bounds__(256) void qkv(const float* __restrict__ kft,
                                           const float* __restrict__ T,
                                           float* __restrict__ X2) {
  int bh  = blockIdx.x >> 1;
  int sub = blockIdx.x & 1;
  const float* K   = kft + (size_t)bh * FT_STRIDE;
  const float* Tre = T + (size_t)bh * 8192;
  const float* Tim = Tre + 4096;
  float* X = X2 + (size_t)bh * FT_STRIDE;

  int tid = threadIdx.x, lane = tid & 31, wave = tid >> 5;
  int mrow = lane & 15, half = lane >> 4;

  int tile = wave + 8 * sub;
  int ei = tile >> 2, xi = tile & 3;
  v8f accR, accI;
#pragma unroll
  for (int r = 0; r < 8; ++r) { accR[r] = 0.0f; accI[r] = 0.0f; }

  for (int y0 = 0; y0 < 64; y0 += 4) {
    v2f aR, aI, aIn, bR, bI;
#pragma unroll
    for (int j = 0; j < 2; ++j) {
      int ky = y0 + j + 2 * half;
      aR[j] = K[(ei * 16 + mrow) * MODES2 + ky];
      aI[j] = K[(ei * 16 + mrow) * MODES2 + 64 + ky];
      bR[j] = Tre[(xi * 16 + mrow) * 64 + ky];   // B[k=y][n=x] = T[x][y]
      bI[j] = Tim[(xi * 16 + mrow) * 64 + ky];
      aIn[j] = -aI[j];
    }
    accR = wmma_f32(aR,  bR, accR);
    accR = wmma_f32(aIn, bI, accR);
    accI = wmma_f32(aR,  bI, accI);
    accI = wmma_f32(aI,  bR, accI);
  }
#pragma unroll
  for (int r = 0; r < 8; ++r) {
    int e = ei * 16 + r + 8 * half;
    int x = xi * 16 + mrow;
    X[e * MODES2 + x]      = accR[r];
    X[e * MODES2 + 64 + x] = accI[r];
  }
}

// ---------------------------------------------------------------------------
// Inverse DFT. grid.x = BH*16, each WG: [64e x 128t] block, K=128 in 4
// double-buffered chunks. Epilogue transposes C through LDS (reusing staging
// buffers) so the 64 MB output is written as contiguous 256B rows.
// ---------------------------------------------------------------------------
__global__ __launch_bounds__(256) void dft_inv(const float* __restrict__ X2,
                                               const __bf16* __restrict__ WinvT,
                                               float* __restrict__ out) {
  __shared__ __align__(16) char smem[37888];
  __bf16* Xs = (__bf16*)smem;                 // [64][136] bf16 = 17408 B
  __bf16* Bs0 = (__bf16*)(smem + 17408);      // [128][40] bf16 = 10240 B
  __bf16* Bs1 = (__bf16*)(smem + 27648);      // [128][40] bf16 = 10240 B
  float*  Cs = (float*)smem;                  // [128][68] f32  = 34816 B (reuse)

  int wg = blockIdx.x;
  int bh = wg >> 4, tblk = wg & 15;
  int b = bh >> 3, h = bh & 7;
  int t0 = tblk * 128;
  const float* Xg = X2 + (size_t)bh * FT_STRIDE;

  int tid = threadIdx.x, lane = tid & 31, wave = tid >> 5;
  int mrow = lane & 15, half = lane >> 4;
  int rt = wave & 3;            // 4 e row tiles, 1 per wave
  int c0 = (wave >> 2) * 4;     // 8 t col tiles, 4 per wave

  int rowB[2], cqB[2];
#pragma unroll
  for (int v = 0; v < 2; ++v) {
    int lin = v * 256 + tid;    // 512 uint4 per chunk
    rowB[v] = lin >> 2;
    cqB[v]  = lin & 3;
  }

  // Stage the whole X tile (f32 -> bf16), coalesced float4 reads.
#pragma unroll
  for (int v = 0; v < 8; ++v) {
    int lin = v * 256 + tid;    // 2048 float4
    int e  = lin >> 5;
    int cq = lin & 31;
    const float4 f = *(const float4*)(Xg + e * MODES2 + cq * 4);
    Xs[e * 136 + cq * 4 + 0] = f2bf(f.x);
    Xs[e * 136 + cq * 4 + 1] = f2bf(f.y);
    Xs[e * 136 + cq * 4 + 2] = f2bf(f.z);
    Xs[e * 136 + cq * 4 + 3] = f2bf(f.w);
  }

  v8f acc[4];
#pragma unroll
  for (int cj = 0; cj < 4; ++cj)
#pragma unroll
    for (int r = 0; r < 8; ++r) acc[cj][r] = 0.0f;

  uint4 fb[2];
  // Prologue: chunk 0 into Bs0.
#pragma unroll
  for (int v = 0; v < 2; ++v)
    fb[v] = *(const uint4*)(WinvT + (size_t)(t0 + rowB[v]) * MODES2 + cqB[v] * 8);
#pragma unroll
  for (int v = 0; v < 2; ++v)
    *(uint4*)(&Bs0[rowB[v] * 40 + cqB[v] * 8]) = fb[v];
  __syncthreads();

#pragma unroll 1
  for (int kc = 0; kc < 4; ++kc) {
    int k0 = kc * 32;
    __bf16* cur = (kc & 1) ? Bs1 : Bs0;
    __bf16* nxt = (kc & 1) ? Bs0 : Bs1;
    bool more = (kc + 1) < 4;
    if (more) {
#pragma unroll
      for (int v = 0; v < 2; ++v)
        fb[v] = *(const uint4*)(WinvT + (size_t)(t0 + rowB[v]) * MODES2 +
                                k0 + 32 + cqB[v] * 8);
    }
    v16bf a;
#pragma unroll
    for (int i = 0; i < 16; ++i)
      a[i] = Xs[(rt * 16 + mrow) * 136 + k0 + a16x32_k(i, half)];
#pragma unroll
    for (int cj = 0; cj < 4; ++cj) {
      v16bf bfrag;
#pragma unroll
      for (int i = 0; i < 16; ++i)
        bfrag[i] = cur[((c0 + cj) * 16 + mrow) * 40 + a16x32_k(i, half)];
      acc[cj] = wmma_bf16(a, bfrag, acc[cj]);
    }
    if (more) {
#pragma unroll
      for (int v = 0; v < 2; ++v)
        *(uint4*)(&nxt[rowB[v] * 40 + cqB[v] * 8]) = fb[v];
    }
    __syncthreads();
  }

  // Transpose C tiles through LDS, then fully-coalesced float4 stores.
#pragma unroll
  for (int cj = 0; cj < 4; ++cj) {
    int tl = (c0 + cj) * 16 + mrow;
#pragma unroll
    for (int r = 0; r < 8; ++r) {
      int e = rt * 16 + r + 8 * half;
      Cs[tl * 68 + e] = acc[cj][r];
    }
  }
  __syncthreads();
  {
    int row  = tid >> 1;                 // 128 t rows, 2 threads per row
    int colb = (tid & 1) * 32;
    int t = t0 + row;
    float* orow = out + (((size_t)b * SEQ + t) * HEADS + h) * EDIM + colb;
    const float* crow = Cs + row * 68 + colb;
#pragma unroll
    for (int j = 0; j < 8; ++j)
      *(float4*)(orow + j * 4) = *(const float4*)(crow + j * 4);
  }
}

// ---------------------------------------------------------------------------
// ws layout (bytes):
//   WfwdT bf16 [128][2048]    @ 0         (524288)
//   WinvT bf16 [2048][128]    @ 524288    (524288)
//   xq_ft f32 [128][64][128]  @ 1048576   (4194304)
//   xk_ft f32 [128][64][128]  @ 5242880   (4194304)  (contiguous after xq_ft)
//   T     f32 [128][2][64][64]@ 9437184   (4194304)
//   X2    f32 [128][64][128]  @ 13631488  (4194304)
// total 17825792 bytes
// ---------------------------------------------------------------------------
extern "C" void kernel_launch(void* const* d_in, const int* in_sizes, int n_in,
                              void* d_out, int out_size, void* d_ws, size_t ws_size,
                              hipStream_t stream) {
  const float* q = (const float*)d_in[0];
  const float* k = (const float*)d_in[1];
  // d_in[2] (v) is unused by the reference.
  char* ws = (char*)d_ws;
  __bf16* WfwdT = (__bf16*)(ws);
  __bf16* WinvT = (__bf16*)(ws + 524288);
  float*  xqft  = (float*)(ws + 1048576);
  float*  T     = (float*)(ws + 9437184);
  float*  X2    = (float*)(ws + 13631488);
  float*  xkft  = xqft + (size_t)BH * FT_STRIDE;
  float*  out   = (float*)d_out;

  twiddle_init<<<1024, 256, 0, stream>>>(WfwdT, WinvT);
  dft_fwd<<<256, 256, 0, stream>>>(q, k, WfwdT, xqft);   // writes xqft then xkft
  qk_tanh<<<2 * BH, 256, 0, stream>>>(xqft, xkft, T);
  qkv<<<2 * BH, 256, 0, stream>>>(xkft, T, X2);
  dft_inv<<<BH * 16, 256, 0, stream>>>(X2, WinvT, out);
}
